// DPTreeMultiheadAttention_21079699489136
// MI455X (gfx1250) — matrix operand order repaired and checked by reading, not compiled, not run on hardware
//
#include <hip/hip_runtime.h>
#include <hip/hip_bf16.h>

// ---------------------------------------------------------------------------
// DP-tree multihead attention for MI455X (gfx1250), wave32 + WMMA bf16 path.
//
// Math reduction: the flip/cumsum DP over the T*T span matrix is linear in k
// and tree spans are laminar, so scores[.,q,n] = <q, Ksub[n]> where Ksub[n]
// is the subtree key-sum of node n. Preorder node order => subtree(n) is the
// contiguous range [n, n+sz_n), so Ksub comes from a suffix sum.
//
// All GEMMs run on v_wmma_f32_16x16x32_bf16 with fp32 accumulation.
// ---------------------------------------------------------------------------

typedef __bf16 bf16;
typedef __attribute__((ext_vector_type(16))) __bf16 v16bf;
typedef __attribute__((ext_vector_type(8)))  float  v8f;

namespace {
constexpr int B_  = 4;
constexpr int H_  = 8;
constexpr int D_  = 128;
constexpr int E_  = 1024;
constexpr int LQ  = 128;
constexpr int TK  = 255;   // tree nodes
constexpr int TKP = 256;   // padded
}

// ------------------------------ fp32 -> bf16 -------------------------------
__global__ void f32_to_bf16_kernel(const float* __restrict__ in,
                                   bf16* __restrict__ out, int n) {
  int i = blockIdx.x * blockDim.x + threadIdx.x;
  int stride = gridDim.x * blockDim.x;
  for (; i < n; i += stride) out[i] = (bf16)in[i];
}

// ------------------------------ epilogues ----------------------------------
// MODE 0: q proj  -> bf16 q[b][h][t][d], (acc+bias)*scale
// MODE 1: k proj  -> f32  k[b][h][node(pad256)][d], acc+bias
// MODE 2: v proj  -> bf16 vT[b][h][d][node(pad256)], acc+bias
// MODE 3: scores  -> f32  scores[z][t][node(pad256)]
// MODE 4: attn    -> bf16 attn[t][b][h*128+n]
// MODE 5: out proj-> f32  d_out[r][n], acc+bias
template <int MODE>
__device__ inline void epilogue_store(void* Out, const float* bias, float val,
                                      int r, int n, int z, int M, float scale) {
  if (r >= M) return;
  if constexpr (MODE == 0) {
    int t = r >> 2, b = r & 3, h = n >> 7, d = n & 127;
    val = (val + bias[n]) * scale;
    ((bf16*)Out)[(((b * H_ + h) * LQ + t) << 7) + d] = (bf16)val;
  } else if constexpr (MODE == 1) {
    int node = r >> 2, b = r & 3, h = n >> 7, d = n & 127;
    val += bias[n];
    ((float*)Out)[(((b * H_ + h) * TKP + node) << 7) + d] = val;
  } else if constexpr (MODE == 2) {
    int node = r >> 2, b = r & 3, h = n >> 7, d = n & 127;
    val += bias[n];
    ((bf16*)Out)[(((b * H_ + h) * D_ + d) << 8) + node] = (bf16)val;
  } else if constexpr (MODE == 3) {
    ((float*)Out)[((long)z * LQ + r) * TKP + n] = val;
  } else if constexpr (MODE == 4) {
    int b = z >> 3, h = z & 7;
    ((bf16*)Out)[((r * B_ + b) << 10) + h * D_ + n] = (bf16)val;
  } else {
    val += bias[n];
    ((float*)Out)[(r << 10) + n] = val;
  }
}

// ------------------------- batched WMMA GEMM -------------------------------
// C[z] = A[z] (MxK, row major, bf16) * BT[z]^T  (BT is NxK row major, bf16)
// 128 threads = 4 waves; workgroup tile 64x64; wave tile 32x32 (2x2 WMMA).
union FragU { v16bf v; uint4 q[2]; };

template <int MODE>
__global__ __launch_bounds__(128)
void wmma_gemm_kernel(const bf16* __restrict__ A, const bf16* __restrict__ BT,
                      void* Out, const float* __restrict__ bias,
                      int M, int N, int K,
                      long aBatchStride, long bBatchStride, float scale) {
  __shared__ bf16 As[64][32];
  __shared__ bf16 Bs[64][32];

  const int tid   = threadIdx.x;
  const int lane  = tid & 31;
  const int wave  = tid >> 5;
  const int waveM = wave >> 1, waveN = wave & 1;
  const int z     = blockIdx.z;
  const int tileM = blockIdx.y * 64;
  const int tileN = blockIdx.x * 64;

  A  += (long)z * aBatchStride;
  BT += (long)z * bBatchStride;

  v8f acc[2][2] = {};

  const int ldRow = tid >> 1;        // 0..63
  const int ldSeg = (tid & 1) * 16;  // 0 or 16 elements (32B chunks)

  const int row16 = lane & 15;
  const int hi    = lane >> 4;   // 0/1
  const int b0    = hi * 8;      // A-frag K base (elems)
  const int kb    = hi * 16;     // unused; B-frag also uses b0

  for (int kk = 0; kk < K; kk += 32) {
    // global -> LDS staging (each thread moves 32B of A and 32B of B)
    {
      int gr = tileM + ldRow;
      uint4 x0 = {}, x1 = {};
      if (gr < M) {
        const uint4* p = (const uint4*)(A + (long)gr * K + kk + ldSeg);
        x0 = p[0];
        x1 = p[1];
      }
      *(uint4*)(&As[ldRow][ldSeg])     = x0;
      *(uint4*)(&As[ldRow][ldSeg + 8]) = x1;

      int gn = tileN + ldRow;  // N is always a multiple of 64 here
      const uint4* pb = (const uint4*)(BT + (long)gn * K + kk + ldSeg);
      uint4 y0 = pb[0];
      uint4 y1 = pb[1];
      *(uint4*)(&Bs[ldRow][ldSeg])     = y0;
      *(uint4*)(&Bs[ldRow][ldSeg + 8]) = y1;
    }
    __syncthreads();

    // A-frag (16x32 bf16): lane holds row lane&15, K chunks [b0,b0+8) and
    // [b0+16,b0+24). B-frag (32x16): lane holds col lane&15, same K chunks
    // [b0,b0+8) and [b0+16,b0+24).
    FragU afr[2], bfr[2];
#pragma unroll
    for (int i = 0; i < 2; ++i) {
      int rA = waveM * 32 + i * 16 + row16;
      afr[i].q[0] = *(const uint4*)(&As[rA][b0]);
      afr[i].q[1] = *(const uint4*)(&As[rA][b0 + 16]);
      int rB = waveN * 32 + i * 16 + row16;
      bfr[i].q[0] = *(const uint4*)(&Bs[rB][b0]);
      bfr[i].q[1] = *(const uint4*)(&Bs[rB][b0 + 16]);
    }
#pragma unroll
    for (int i = 0; i < 2; ++i)
#pragma unroll
      for (int j = 0; j < 2; ++j)
        acc[i][j] = __builtin_amdgcn_wmma_f32_16x16x32_bf16(
            false, afr[i].v, false, bfr[j].v, (short)0, acc[i][j],
            false, false);
    __syncthreads();
  }

  // C layout: VGPR e, lane l -> row e + 8*(l>>4), col l&15 (per 16x16 tile)
#pragma unroll
  for (int i = 0; i < 2; ++i) {
#pragma unroll
    for (int j = 0; j < 2; ++j) {
      int baseR = tileM + waveM * 32 + i * 16 + hi * 8;
      int baseC = tileN + waveN * 32 + j * 16 + row16;
#pragma unroll
      for (int e = 0; e < 8; ++e)
        epilogue_store<MODE>(Out, bias, acc[i][j][e], baseR + e, baseC, z, M,
                             scale);
    }
  }
}

// --------------------- subtree key-sums via suffix sums --------------------
// Preorder tree: subtree(n) = [n, n+sz_n), sz_n = 2*(hi-lo+1)-1.
// Ksub[n][d] = suffix[n][d] - suffix[n+sz_n][d].
__global__ __launch_bounds__(128)
void subtree_sum_kernel(const float* __restrict__ kf, const int* __restrict__ idx,
                        float* __restrict__ suffix, bf16* __restrict__ ksub) {
  const int bh = blockIdx.x;       // 0..31 (b*8+h)
  const int d  = threadIdx.x;      // 0..127
  const int b  = bh >> 3;

  const float* kc = kf + (long)bh * TKP * D_;
  float* sf = suffix + (long)bh * TKP * D_;
  bf16*  ko = ksub + (long)bh * TKP * D_;

  sf[TK * D_ + d] = 0.0f;
  float acc = 0.0f;
  for (int n = TK - 1; n >= 0; --n) {
    acc += kc[n * D_ + d];
    sf[n * D_ + d] = acc;
  }
  for (int n = 0; n < TK; ++n) {
    int lo  = idx[(b * TK + n) * 2 + 0];
    int hi2 = idx[(b * TK + n) * 2 + 1];
    int sz  = 2 * (hi2 - lo + 1) - 1;
    float v = sf[n * D_ + d] - sf[(n + sz) * D_ + d];
    ko[n * D_ + d] = (bf16)v;
  }
  ko[TK * D_ + d] = (bf16)0.0f;  // zero pad row -> pad scores are exactly 0
}

// ------------------------------- softmax -----------------------------------
__global__ __launch_bounds__(128)
void softmax_kernel(const float* __restrict__ scores, bf16* __restrict__ w) {
  const int bh = blockIdx.x;
  const int t  = threadIdx.x;
  const float* s = scores + ((long)bh * LQ + t) * TKP;
  bf16* o = w + ((long)bh * LQ + t) * TKP;

  float m = -3.4e38f;
  for (int n = 0; n < TK; ++n) m = fmaxf(m, s[n]);
  float sum = 0.0f;
  for (int n = 0; n < TK; ++n) sum += __expf(s[n] - m);
  float inv = 1.0f / sum;
  for (int n = 0; n < TK; ++n) o[n] = (bf16)(__expf(s[n] - m) * inv);
  o[TK] = (bf16)0.0f;  // pad column contributes nothing to attn*V
}

// ------------------------------- launcher ----------------------------------
extern "C" void kernel_launch(void* const* d_in, const int* in_sizes, int n_in,
                              void* d_out, int out_size, void* d_ws,
                              size_t ws_size, hipStream_t stream) {
  const float* query  = (const float*)d_in[0];  // [128,4,1024]
  const float* key    = (const float*)d_in[1];  // [255,4,1024]
  const float* value  = (const float*)d_in[2];  // [255,4,1024]
  const int*   indices = (const int*)d_in[3];   // [4,255,2]
  const float* in_w   = (const float*)d_in[4];  // [3072,1024]
  const float* in_b   = (const float*)d_in[5];  // [3072]
  const float* out_w  = (const float*)d_in[6];  // [1024,1024]
  const float* out_b  = (const float*)d_in[7];  // [1024]

  char* ws = (char*)d_ws;
  auto carve = [&](size_t bytes) {
    char* p = ws;
    ws += (bytes + 255) & ~(size_t)255;
    return p;
  };
  bf16*  query_bf = (bf16*)carve((size_t)512 * 1024 * 2);
  bf16*  key_bf   = (bf16*)carve((size_t)1020 * 1024 * 2);
  bf16*  value_bf = (bf16*)carve((size_t)1020 * 1024 * 2);
  bf16*  win_bf   = (bf16*)carve((size_t)3072 * 1024 * 2);
  bf16*  wo_bf    = (bf16*)carve((size_t)1024 * 1024 * 2);
  bf16*  q_bf     = (bf16*)carve((size_t)32 * 128 * 128 * 2);
  float* k_f      = (float*)carve((size_t)32 * 256 * 128 * 4);
  bf16*  vT_bf    = (bf16*)carve((size_t)32 * 128 * 256 * 2);
  float* suffix   = (float*)carve((size_t)32 * 256 * 128 * 4);
  bf16*  ksub_bf  = (bf16*)carve((size_t)32 * 256 * 128 * 2);
  float* scores   = (float*)carve((size_t)32 * 128 * 256 * 4);
  bf16*  wsm_bf   = (bf16*)carve((size_t)32 * 128 * 256 * 2);
  bf16*  attn_bf  = (bf16*)carve((size_t)128 * 4 * 1024 * 2);
  (void)ws_size; (void)in_sizes; (void)n_in; (void)out_size;

  const float qscale = 0.08838834764831845f;  // 128^-0.5

  // 1) fp32 -> bf16 conversions
  auto cvt = [&](const float* src, bf16* dst, int n) {
    f32_to_bf16_kernel<<<(n + 255) / 256, 256, 0, stream>>>(src, dst, n);
  };
  cvt(query, query_bf, 512 * 1024);
  cvt(key, key_bf, 1020 * 1024);
  cvt(value, value_bf, 1020 * 1024);
  cvt(in_w, win_bf, 3072 * 1024);
  cvt(out_w, wo_bf, 1024 * 1024);

  // 2) projections  (C = X * W^T; W slices of in_proj are already [N,K])
  wmma_gemm_kernel<0><<<dim3(16, 8, 1), 128, 0, stream>>>(
      query_bf, win_bf, q_bf, in_b, 512, 1024, 1024, 0, 0, qscale);
  wmma_gemm_kernel<1><<<dim3(16, 16, 1), 128, 0, stream>>>(
      key_bf, win_bf + (size_t)1024 * 1024, k_f, in_b + 1024, 1020, 1024, 1024,
      0, 0, 1.0f);
  wmma_gemm_kernel<2><<<dim3(16, 16, 1), 128, 0, stream>>>(
      value_bf, win_bf + (size_t)2048 * 1024, vT_bf, in_b + 2048, 1020, 1024,
      1024, 0, 0, 1.0f);

  // 3) subtree key sums (suffix-sum trick)
  subtree_sum_kernel<<<32, 128, 0, stream>>>(k_f, indices, suffix, ksub_bf);

  // 4) scores[z] = q[z] * Ksub[z]^T   (z = b*8+h)
  wmma_gemm_kernel<3><<<dim3(4, 2, 32), 128, 0, stream>>>(
      q_bf, ksub_bf, scores, nullptr, 128, 256, 128, (long)128 * 128,
      (long)256 * 128, 1.0f);

  // 5) softmax over 255 nodes, bf16 weights with zero pad column
  softmax_kernel<<<32, 128, 0, stream>>>(scores, wsm_bf);

  // 6) attn[z] = W[z] * V[z]   (vT layout makes B-fragments contiguous)
  wmma_gemm_kernel<4><<<dim3(2, 2, 32), 128, 0, stream>>>(
      wsm_bf, vT_bf, attn_bf, nullptr, 128, 128, 256, (long)128 * 256,
      (long)128 * 256, 1.0f);

  // 7) output projection + bias -> d_out  [128,4,1024] f32
  wmma_gemm_kernel<5><<<dim3(16, 8, 1), 128, 0, stream>>>(
      attn_bf, wo_bf, d_out, out_b, 512, 1024, 1024, 0, 0, 1.0f);
}
